// WaveNet_5033701671460
// MI455X (gfx1250) — matrix operand order repaired and checked by reading, not compiled, not run on hardware
//
#include <hip/hip_runtime.h>

typedef __attribute__((ext_vector_type(16))) _Float16 v16h;
typedef __attribute__((ext_vector_type(8)))  _Float16 v8h;
typedef __attribute__((ext_vector_type(8)))  float    v8f;
typedef __attribute__((address_space(3))) unsigned char lds_u8;

namespace {
constexpr int  Bn = 16, Hc = 256, T = 4096;
constexpr int  R = 128, S = 256, L = 16, NC = 12;
constexpr long N = (long)Bn * T;          // 65536 columns
constexpr int  NBLK = 64;                 // columns per block tile
constexpr int  LDSP = 8;                  // LDS row pad (halves) -> bank-conflict-free
constexpr long D_IN   = (long)R * Hc;                 // 32768
constexpr long D_DIL  = (long)L * 2 * (2 * R) * R;    // 1048576
constexpr long D_RES  = (long)(L - 1) * R * R;        // 245760
constexpr long D_SKIP = (long)L * S * R;              // 524288
constexpr long WTOT   = D_IN + D_DIL + D_RES + D_SKIP;
}

// ---------------- WMMA helpers (gfx1250 wave32 layouts) ----------------

__device__ inline v8f wmma_f16(v16h a, v16h b, v8f c) {
  return __builtin_amdgcn_wmma_f32_16x16x32_f16(
      /*neg_a=*/false, a, /*neg_b=*/false, b,
      /*c_mod=*/(short)0, c, /*reuse_a=*/false, /*reuse_b=*/false);
}

// A fragment 16x32: Wmk points at (row m_base, col k_base), row-major, stride ldk.
// lanes 0-15: rows, K 0-15; lanes 16-31: rows, K 16-31.
__device__ inline v16h a_frag(const _Float16* __restrict__ Wmk, int ldk, int lane) {
  const int hq = lane >> 4, m = lane & 15;
  const _Float16* p = Wmk + (long)m * ldk + hq * 16;
  v16h a;
#pragma unroll
  for (int j = 0; j < 8; ++j) a[j] = p[j];
#pragma unroll
  for (int j = 0; j < 8; ++j) a[8 + j] = p[8 + j];
  return a;
}

// B fragment 32x16 from LDS, activations stored [n][k]:
// per lane: fixed column n, K = (lane/16)*16 + j  -> 16 contiguous halves.
__device__ inline v16h b_frag_lds(const _Float16* p) {
  v16h b;
#pragma unroll
  for (int j = 0; j < 16; ++j) b[j] = p[j];
  return b;
}

// ---------------- async staging: global -> LDS direct path (ASYNCcnt) ----------

// 16B per-lane DMA into LDS, bypassing VGPRs (CDNA5 ISA).
__device__ inline void async_cp16(_Float16* lp, const _Float16* gp) {
  asm volatile("global_load_async_to_lds_b128 %0, %1, off"
               :
               : "v"((lds_u8*)lp), "v"((unsigned long long)(uintptr_t)gp)
               : "memory");
}

__device__ inline void async_wait0() {
  asm volatile("s_wait_asynccnt 0x0" ::: "memory");
}

template <int K>
__device__ inline void stage_tile(_Float16* __restrict__ lds,
                                  const _Float16* __restrict__ src, long n0) {
  constexpr int CPR = K / 8;            // 16B chunks per row
  constexpr int TOT = NBLK * CPR;
  for (int idx = threadIdx.x; idx < TOT; idx += 256) {
    int r = idx / CPR, c = idx % CPR;
    async_cp16(lds + r * (K + LDSP) + c * 8, src + (n0 + r) * K + (long)c * 8);
  }
}

// Causal-shifted tile: row r holds x[n-d]; zero where (n % T) < d.
template <int K>
__device__ inline void stage_tile_shift(_Float16* __restrict__ lds,
                                        const _Float16* __restrict__ src,
                                        long n0, int d) {
  constexpr int CPR = K / 8;
  constexpr int TOT = NBLK * CPR;
  for (int idx = threadIdx.x; idx < TOT; idx += 256) {
    int r = idx / CPR, c = idx % CPR;
    long n = n0 + r;
    int  t = (int)(n & (T - 1));
    if (t >= d) {
      async_cp16(lds + r * (K + LDSP) + c * 8, src + (n - d) * K + (long)c * 8);
    } else {
      *(uint4*)(lds + r * (K + LDSP) + c * 8) = make_uint4(0u, 0u, 0u, 0u);
    }
  }
}

// ---------------- data preparation ----------------

// input [B][H][T] f32 -> [N][H] f16 (N = b*T + t)
__global__ __launch_bounds__(256) void k_transpose_in(
    const float* __restrict__ in, _Float16* __restrict__ inh) {
  long o = (long)blockIdx.x * 256 + threadIdx.x;   // over N*Hc
  if (o >= N * Hc) return;
  long n = o / Hc; int h = (int)(o % Hc);
  long b = n / T, t = n % T;
  inh[o] = (_Float16)in[(b * Hc + h) * T + t];
}

// weights f32 -> f16; dilate reordered [L][tap][2R][R]
__global__ __launch_bounds__(256) void k_wconv(
    const float* __restrict__ w_in, const float* __restrict__ dil_w,
    const float* __restrict__ res_w, const float* __restrict__ skip_w,
    _Float16* __restrict__ wbuf) {
  long gid = (long)blockIdx.x * 256 + threadIdx.x;
  if (gid < D_IN) {
    wbuf[gid] = (_Float16)w_in[gid];
  } else if (gid < D_IN + D_DIL) {
    long loc = gid - D_IN;
    long per = (long)2 * (2 * R) * R;           // per layer
    long l = loc / per, r1 = loc % per;
    long k = r1 / ((long)(2 * R) * R);
    long r2 = r1 % ((long)(2 * R) * R);
    long o = r2 / R, i = r2 % R;
    wbuf[gid] = (_Float16)dil_w[((l * (2 * R) + o) * R + i) * 2 + k];
  } else if (gid < D_IN + D_DIL + D_RES) {
    wbuf[gid] = (_Float16)res_w[gid - (D_IN + D_DIL)];
  } else if (gid < WTOT) {
    wbuf[gid] = (_Float16)skip_w[gid - (D_IN + D_DIL + D_RES)];
  }
}

// ---------------- GEMM kernels ----------------

// x[n][c] = Win[c][:] . in[n][:] + b_in[c]    (M=128, K=256)
__global__ __launch_bounds__(256) void k_inconv(
    const _Float16* __restrict__ inh, const _Float16* __restrict__ winh,
    const float* __restrict__ bin, float* __restrict__ x,
    _Float16* __restrict__ xh) {
  __shared__ _Float16 lds[NBLK * (Hc + LDSP)];
  const long n0 = (long)blockIdx.x * NBLK;
  stage_tile<Hc>(lds, inh, n0);
  async_wait0();
  __syncthreads();
  const int lane = threadIdx.x & 31, w = threadIdx.x >> 5;
  const int hq = lane >> 4, nl = lane & 15;
  v8f acc[4] = {};
  for (int kb = 0; kb < Hc; kb += 32) {
    v16h a = a_frag(winh + (long)(16 * w) * Hc + kb, Hc, lane);
#pragma unroll
    for (int nt = 0; nt < 4; ++nt) {
      v16h b = b_frag_lds(&lds[(nt * 16 + nl) * (Hc + LDSP) + kb + hq * 16]);
      acc[nt] = wmma_f16(a, b, acc[nt]);
    }
  }
  const int mb = 16 * w + 8 * hq;
#pragma unroll
  for (int nt = 0; nt < 4; ++nt) {
    long n = n0 + nt * 16 + nl;
    float* xp = x + n * R + mb;
    _Float16* xhp = xh + n * R + mb;
#pragma unroll
    for (int i = 0; i < 8; ++i) {
      float v = acc[nt][i] + bin[mb + i];
      xp[i] = v;
      xhp[i] = (_Float16)v;
    }
  }
}

// a = W0 @ x(t-d) + W1 @ x(t) + b ; acts = tanh(a_lo) * sigmoid(a_hi)
// M = 2R = 256 (wave w owns rows 16w (tanh) and 16w+128 (sigmoid)), K = 128 per tap.
__global__ __launch_bounds__(256) void k_dilate_gate(
    const _Float16* __restrict__ xh, const _Float16* __restrict__ wdil,
    const float* __restrict__ bias, _Float16* __restrict__ acts, int d) {
  __shared__ _Float16 lds[2][NBLK * (R + LDSP)];
  const long n0 = (long)blockIdx.x * NBLK;
  stage_tile_shift<R>(lds[0], xh, n0, d);   // tap 0: x[t-d]
  stage_tile<R>(lds[1], xh, n0);            // tap 1: x[t]
  async_wait0();
  __syncthreads();
  const int lane = threadIdx.x & 31, w = threadIdx.x >> 5;
  const int hq = lane >> 4, nl = lane & 15;
  v8f accT[4] = {}, accS[4] = {};
  for (int kb = 0; kb < R; kb += 32) {
#pragma unroll
    for (int tap = 0; tap < 2; ++tap) {
      const _Float16* W = wdil + (long)tap * (2 * R) * R + kb;
      v16h aT = a_frag(W + (long)(16 * w) * R, R, lane);
      v16h aS = a_frag(W + (long)(16 * w + R) * R, R, lane);
#pragma unroll
      for (int nt = 0; nt < 4; ++nt) {
        v16h b = b_frag_lds(&lds[tap][(nt * 16 + nl) * (R + LDSP) + kb + hq * 16]);
        accT[nt] = wmma_f16(aT, b, accT[nt]);
        accS[nt] = wmma_f16(aS, b, accS[nt]);
      }
    }
  }
  const int mb = 16 * w + 8 * hq;
  float bT[8], bS[8];
#pragma unroll
  for (int i = 0; i < 8; ++i) { bT[i] = bias[mb + i]; bS[i] = bias[R + mb + i]; }
#pragma unroll
  for (int nt = 0; nt < 4; ++nt) {
    long n = n0 + nt * 16 + nl;
    v8h o;
#pragma unroll
    for (int i = 0; i < 8; ++i) {
      float tv = tanhf(accT[nt][i] + bT[i]);
      float sv = accS[nt][i] + bS[i];
      sv = 1.0f / (1.0f + __expf(-sv));
      o[i] = (_Float16)(tv * sv);
    }
    *(v8h*)(acts + n * R + mb) = o;
  }
}

// Stacked res (M=128, adds residual -> x f32+f16) + skip (M=256, accumulates).
template <bool HAS_RES, bool ACCUM>
__global__ __launch_bounds__(256) void k_res_skip(
    const _Float16* __restrict__ acts,
    const _Float16* __restrict__ wres, const float* __restrict__ bres,
    const _Float16* __restrict__ wskp, const float* __restrict__ bskp,
    float* __restrict__ x, _Float16* __restrict__ xh,
    float* __restrict__ skip) {
  constexpr int TPW = HAS_RES ? 3 : 2;   // m-tiles per wave (24 or 16 total)
  __shared__ _Float16 lds[NBLK * (R + LDSP)];
  const long n0 = (long)blockIdx.x * NBLK;
  stage_tile<R>(lds, acts, n0);
  async_wait0();
  __syncthreads();
  const int lane = threadIdx.x & 31, w = threadIdx.x >> 5;
  const int hq = lane >> 4, nl = lane & 15;
  v8f acc[TPW][4] = {};
  for (int kb = 0; kb < R; kb += 32) {
    v16h bf[4];
#pragma unroll
    for (int nt = 0; nt < 4; ++nt)
      bf[nt] = b_frag_lds(&lds[(nt * 16 + nl) * (R + LDSP) + kb + hq * 16]);
#pragma unroll
    for (int j = 0; j < TPW; ++j) {
      int mt = w * TPW + j;
      const _Float16* Wp;
      if (HAS_RES && mt < 8) Wp = wres + (long)(16 * mt) * R + kb;
      else {
        int sm = HAS_RES ? (mt - 8) : mt;
        Wp = wskp + (long)(16 * sm) * R + kb;
      }
      v16h a = a_frag(Wp, R, lane);
#pragma unroll
      for (int nt = 0; nt < 4; ++nt) acc[j][nt] = wmma_f16(a, bf[nt], acc[j][nt]);
    }
  }
#pragma unroll
  for (int j = 0; j < TPW; ++j) {
    int mt = w * TPW + j;
    if (HAS_RES && mt < 8) {
      int mb = 16 * mt + 8 * hq;
#pragma unroll
      for (int nt = 0; nt < 4; ++nt) {
        long n = n0 + nt * 16 + nl;
        float* xp = x + n * R + mb;
        _Float16* xhp = xh + n * R + mb;
#pragma unroll
        for (int i = 0; i < 8; ++i) {
          float v = acc[j][nt][i] + bres[mb + i] + xp[i];
          xp[i] = v;
          xhp[i] = (_Float16)v;
        }
      }
    } else {
      int sm = HAS_RES ? (mt - 8) : mt;
      int mb = 16 * sm + 8 * hq;
#pragma unroll
      for (int nt = 0; nt < 4; ++nt) {
        long n = n0 + nt * 16 + nl;
        float* sp = skip + n * S + mb;
#pragma unroll
        for (int i = 0; i < 8; ++i) {
          float v = acc[j][nt][i] + bskp[mb + i];
          sp[i] = ACCUM ? (sp[i] + v) : v;
        }
      }
    }
  }
}

// ---------------- head: out-conv + mean, then linear ----------------

__global__ __launch_bounds__(256) void k_out(const float* __restrict__ skip,
                                             const float* __restrict__ wout,
                                             float* __restrict__ outc) {
  int b = blockIdx.x / NC, c = blockIdx.x % NC;
  const float* wr = wout + (long)c * S;
  float sum = 0.f;
  for (int t = threadIdx.x; t < T; t += 256) {
    const float* row = skip + ((long)b * T + t) * S;
#pragma unroll 4
    for (int s = 0; s < S; s += 4) {
      float4 v = *(const float4*)(row + s);
      float4 ww = *(const float4*)(wr + s);
      sum += v.x * ww.x + v.y * ww.y + v.z * ww.z + v.w * ww.w;
    }
  }
  __shared__ float red[256];
  red[threadIdx.x] = sum;
  __syncthreads();
  for (int off = 128; off > 0; off >>= 1) {
    if (threadIdx.x < off) red[threadIdx.x] += red[threadIdx.x + off];
    __syncthreads();
  }
  if (threadIdx.x == 0) outc[b * NC + c] = red[0] / (float)T;
}

__global__ __launch_bounds__(256) void k_linear(const float* __restrict__ outc,
                                                const float* __restrict__ lw,
                                                const float* __restrict__ lb,
                                                float* __restrict__ out) {
  int i = threadIdx.x;
  if (i < Bn * NC) {
    int b = i / NC, j = i % NC;
    float s = lb[j];
#pragma unroll
    for (int c = 0; c < NC; ++c) s += outc[b * NC + c] * lw[j * NC + c];
    out[i] = s;
  }
}

// ---------------- launch ----------------

extern "C" void kernel_launch(void* const* d_in, const int* in_sizes, int n_in,
                              void* d_out, int out_size, void* d_ws, size_t ws_size,
                              hipStream_t stream) {
  (void)in_sizes; (void)n_in; (void)out_size; (void)ws_size;
  const float* input    = (const float*)d_in[0];
  const float* w_in     = (const float*)d_in[1];
  const float* b_in     = (const float*)d_in[2];
  const float* dilate_w = (const float*)d_in[3];
  const float* dilate_b = (const float*)d_in[4];
  const float* res_w    = (const float*)d_in[5];
  const float* res_b    = (const float*)d_in[6];
  const float* skip_w   = (const float*)d_in[7];
  const float* skip_b   = (const float*)d_in[8];
  const float* w_out    = (const float*)d_in[9];
  const float* lin_w    = (const float*)d_in[10];
  const float* lin_b    = (const float*)d_in[11];
  float* out = (float*)d_out;

  char* p = (char*)d_ws;
  auto alloc = [&](size_t bytes) {
    char* q = p;
    p += (bytes + 255) & ~(size_t)255;
    return q;
  };
  float*    x    = (float*)   alloc(sizeof(float)    * R  * N);   // 33.5 MB
  float*    skip = (float*)   alloc(sizeof(float)    * S  * N);   // 67 MB
  _Float16* xh   = (_Float16*)alloc(sizeof(_Float16) * R  * N);   // 16.8 MB
  _Float16* acts = (_Float16*)alloc(sizeof(_Float16) * R  * N);   // 16.8 MB
  _Float16* inh  = (_Float16*)alloc(sizeof(_Float16) * Hc * N);   // 33.5 MB
  _Float16* wbuf = (_Float16*)alloc(sizeof(_Float16) * WTOT);     // 3.7 MB
  float*    outc = (float*)   alloc(sizeof(float)    * Bn * NC);

  const _Float16* winh = wbuf;
  const _Float16* dilh = wbuf + D_IN;
  const _Float16* resh = dilh + D_DIL;
  const _Float16* skph = resh + D_RES;

  k_transpose_in<<<(int)((N * Hc) / 256), 256, 0, stream>>>(input, inh);
  k_wconv<<<(int)((WTOT + 255) / 256), 256, 0, stream>>>(w_in, dilate_w, res_w,
                                                         skip_w, wbuf);
  const int gemm_grid = (int)(N / NBLK);   // 1024
  k_inconv<<<gemm_grid, 256, 0, stream>>>(inh, winh, b_in, x, xh);

  for (int l = 0; l < L; ++l) {
    int d = 1 << (l % 9);
    k_dilate_gate<<<gemm_grid, 256, 0, stream>>>(
        xh, dilh + (long)l * (2 * (2 * R) * R), dilate_b + l * (2 * R), acts, d);
    if (l == 0) {
      k_res_skip<true, false><<<gemm_grid, 256, 0, stream>>>(
          acts, resh, res_b, skph, skip_b, x, xh, skip);
    } else if (l < L - 1) {
      k_res_skip<true, true><<<gemm_grid, 256, 0, stream>>>(
          acts, resh + (long)l * R * R, res_b + l * R,
          skph + (long)l * S * R, skip_b + l * S, x, xh, skip);
    } else {
      // last layer: reference's residual update is dead; only skip accumulates
      k_res_skip<false, true><<<gemm_grid, 256, 0, stream>>>(
          acts, nullptr, nullptr, skph + (long)l * S * R, skip_b + l * S, x, xh,
          skip);
    }
  }
  k_out<<<Bn * NC, 256, 0, stream>>>(skip, w_out, outc);
  k_linear<<<1, 256, 0, stream>>>(outc, lin_w, lin_b, out);
}
